// PolynomialLayer_11141145165863
// MI455X (gfx1250) — matrix operand order repaired and checked by reading, not compiled, not run on hardware
//
#include <hip/hip_runtime.h>

// ---- vector types ----
typedef __attribute__((ext_vector_type(16))) _Float16 v16h;
typedef __attribute__((ext_vector_type(8)))  _Float16 v8h;
typedef __attribute__((ext_vector_type(8)))  float    v8f;
typedef __attribute__((ext_vector_type(4)))  float    v4f;
typedef __attribute__((ext_vector_type(4)))  unsigned v4u;

#define BATCH   2048
#define NFEAT   64
#define NOUT    64
#define NB      32            // batch tile per workgroup
#define KSPLIT  4             // K-dimension split across gridDim.y
#define K1      64
#define K2      2080
#define K3      45760
#define KTOT    (K1 + K2 + K3)    // 47904
#define KPAD    47936             // padded to multiple of 64 (zero coeffs, mono==1)
#define NG      (KPAD / 64)       // 749 granules of K=64 (2 WMMA k-steps each)
#define MSTR2   72                // mono LDS row stride in halves (64 + 8 pad, 16B-aligned)

// idxAll[k] packs (a<<14)|(b<<7)|c ; feature index 64 is a dummy "1.0" feature,
// so monomial(k) = x[a]*x[b]*x[c] uniformly for all degrees (and for the padding).
__global__ void poly_setup(unsigned* __restrict__ idxAll) {
    int t = blockIdx.x * 256 + threadIdx.x;
    if (t < K1) {
        idxAll[t] = (64u << 14) | (64u << 7) | (unsigned)t;      // degree 1: x[t]
    }
    if (t < (KPAD - KTOT)) {                                     // zero-coeff padding
        idxAll[KTOT + t] = (64u << 14) | (64u << 7) | 64u;       // mono == 1.0
    }
    if (t < K2) {
        // decode ordered pair (a,b), a<=b<64, lexicographic index t
        int a = 0;
        while (a < 63 && (2080 - (63 - a) * (64 - a) / 2) <= t) ++a;
        int off2a = 2080 - (64 - a) * (65 - a) / 2;
        int b = a + (t - off2a);
        idxAll[K1 + t] = (64u << 14) | ((unsigned)a << 7) | (unsigned)b;  // degree 2
        // degree-3 run: all (a,b,c), c = b..63
        int off3a = 45760 - (66 - a) * (65 - a) * (64 - a) / 6;
        int S64a = (64 - a) * (65 - a) / 2;
        int S64b = (64 - b) * (65 - b) / 2;
        int base = (K1 + K2) + off3a + (S64a - S64b);
        for (int c = b; c < 64; ++c)
            idxAll[base + (c - b)] = ((unsigned)a << 14) | ((unsigned)b << 7) | (unsigned)c;
    }
}

// One-time concat + f32->f16 convert of all coefficient matrices: Ch[o][k], k in [0,KPAD)
__global__ void poly_cvt(const float* __restrict__ c1,
                         const float* __restrict__ c2,
                         const float* __restrict__ c3,
                         _Float16* __restrict__ Ch) {
    int i = blockIdx.x * 256 + threadIdx.x;    // 0 .. 64*KPAD-1
    int o = i / KPAD;
    int k = i - o * KPAD;
    float v;
    if (k < K1)            v = c1[o * K1 + k];
    else if (k < K1 + K2)  v = c2[o * K2 + (k - K1)];
    else if (k < KTOT)     v = c3[(size_t)o * K3 + (k - K1 - K2)];
    else                   v = 0.0f;
    Ch[i] = (_Float16)v;
}

// ys layout: [KSPLIT][NOUT][BATCH] f32 partial results
__global__ __launch_bounds__(256)
void poly_main(const float* __restrict__ x,
               const _Float16* __restrict__ Ch,
               const unsigned* __restrict__ idxAll,
               float* __restrict__ ys) {
    __shared__ float    xs[(NFEAT + 1) * NB];     // [65][32] feature-major x tile (+ones row)
    __shared__ _Float16 mono[2][NB * MSTR2];      // double-buffered monomial tile [32 nb][64 kk]

    const int t     = threadIdx.x;
    const int btile = blockIdx.x * NB;
    const int split = blockIdx.y;

    // load x tile transposed: xs[f][nb] = x[btile+nb][f]
    for (int i = t; i < NFEAT * NB; i += 256) {
        int f = i & 63, nb = i >> 6;
        xs[f * NB + nb] = x[(size_t)(btile + nb) * NFEAT + f];
    }
    if (t < NB) xs[NFEAT * NB + t] = 1.0f;        // dummy feature == 1
    __syncthreads();

    const int w    = t >> 5;          // wave id 0..7 (wave32)
    const int lane = t & 31;
    const int mw   = w & 3;           // out tile   (M = 16*mw .. +16)
    const int nw   = w >> 2;          // batch sub-tile (N = 16*nw .. +16)
    const int hi   = lane >> 4;       // lane half
    const int lm   = lane & 15;
    const int o    = mw * 16 + lm;    // output row handled by this lane

    // mono build assignment: thread -> (nb, 8 consecutive kk)
    const int mb_nb = t >> 3;         // 0..31
    const int mb_kb = (t & 7) * 8;    // 0,8,...,56

    v8f acc = {};
    const int lo  = (split * NG) / KSPLIT;
    const int hiG = ((split + 1) * NG) / KSPLIT;

    const _Float16* rowbase = Ch + (size_t)o * KPAD + hi * 16;

    for (int g = lo; g < hiG; ++g) {
        const int buf = g & 1;
        // ---- cooperative monomial build: 64 K-values x 32 batch (B operand source) ----
        {
            const unsigned* ip = idxAll + g * 64 + mb_kb;
            v4u p0 = *(const v4u*)(ip);
            v4u p1 = *(const v4u*)(ip + 4);
            v8h vals;
#pragma unroll
            for (int u = 0; u < 8; ++u) {
                unsigned p = (u < 4) ? p0[u] : p1[u - 4];
                int a = (p >> 14) & 127, b = (p >> 7) & 127, c = p & 127;
                float v = xs[a * NB + mb_nb] * xs[b * NB + mb_nb] * xs[c * NB + mb_nb];
                vals[u] = (_Float16)v;
            }
            *(v8h*)(&mono[buf][mb_nb * MSTR2 + mb_kb]) = vals;   // one ds_store_b128
        }
        __syncthreads();   // single barrier per granule (double-buffered mono)

        const _Float16* rowh = rowbase + g * 64;
        __builtin_prefetch(rowh + 8 * 64, 0, 3);   // pull future coeffs L2 -> WGP

#pragma unroll
        for (int s = 0; s < 2; ++s) {
            // ---- A fragment (pre-converted f16 coeffs): halves i -> K=s*32+16*hi+i
            //      (two b128 loads, no conversion)
            union { v4u u[2]; v16h h; } Af;
            const v4u* ap = (const v4u*)(rowh + s * 32);
            Af.u[0] = ap[0];
            Af.u[1] = ap[1];

            // ---- B fragment from LDS: lane lm = column N, halves i -> K = s*32 + 16*hi + i
            union { v4f f[2]; v16h h; } Bf;
            const v4f* bp = (const v4f*)(&mono[buf][(nw * 16 + lm) * MSTR2 + s * 32 + hi * 16]);
            Bf.f[0] = bp[0];
            Bf.f[1] = bp[1];

            // ---- D = A*B + C  (f32 accumulate)
            acc = __builtin_amdgcn_wmma_f32_16x16x32_f16(false, Af.h, false, Bf.h,
                                                         (short)0, acc, false, false);
        }
    }

    // ---- write partial result: D layout: VGPR r, lane -> M = r + 8*hi, N = lm
    float* yrow = ys + (size_t)split * (NOUT * BATCH);
    const int bcol = btile + nw * 16 + lm;
#pragma unroll
    for (int r = 0; r < 8; ++r) {
        int oo = mw * 16 + r + 8 * hi;
        yrow[(size_t)oo * BATCH + bcol] = acc[r];
    }
}

// sum K-splits, add c0, apply the reference's reshape/transpose quirk:
// d_out[(k%64)*2048 + (k/64)] = y_flat[k],  y row-major [64][2048]
__global__ void poly_fin(const float* __restrict__ ys,
                         const float* __restrict__ c0,
                         float* __restrict__ out) {
    int k = blockIdx.x * 256 + threadIdx.x;     // 0 .. 131071
    float v = c0[k >> 11];                      // o = k / 2048
    for (int s = 0; s < KSPLIT; ++s) v += ys[(size_t)s * (NOUT * BATCH) + k];
    out[(size_t)(k & 63) * BATCH + (k >> 6)] = v;
}

extern "C" void kernel_launch(void* const* d_in, const int* in_sizes, int n_in,
                              void* d_out, int out_size, void* d_ws, size_t ws_size,
                              hipStream_t stream) {
    const float* inp = (const float*)d_in[0];   // [2048,64]
    const float* c0  = (const float*)d_in[1];   // [64,1]
    const float* c1  = (const float*)d_in[2];   // [64,64]
    const float* c2  = (const float*)d_in[3];   // [64,2080]
    const float* c3  = (const float*)d_in[4];   // [64,45760]

    // workspace layout
    unsigned*  idxAll = (unsigned*)d_ws;                              // KPAD u32   (187.3 KB)
    _Float16*  Ch     = (_Float16*)((char*)d_ws + 196608);            // 64*KPAD f16 (5.85 MB)
    float*     ys     = (float*)((char*)d_ws + 6332416);              // KSPLIT*64*2048 f32 (2 MB)

    poly_setup<<<9, 256, 0, stream>>>(idxAll);
    poly_cvt<<<(NOUT * KPAD) / 256, 256, 0, stream>>>(c1, c2, c3, Ch);
    poly_main<<<dim3(BATCH / NB, KSPLIT), 256, 0, stream>>>(inp, Ch, idxAll, ys);
    poly_fin<<<(NOUT * BATCH) / 256, 256, 0, stream>>>(ys, c0, (float*)d_out);
}
